// QuantumLayer_11931419148828
// MI455X (gfx1250) — matrix operand order repaired and checked by reading, not compiled, not run on hardware
//
#include <hip/hip_runtime.h>
#include <hip/hip_bf16.h>
#include <math.h>

// MI455X / gfx1250, wave32. One wave simulates one batch element with the
// 1024-amplitude statevector held entirely in VGPRs (32 complex amps/lane).
// Qubits 0-4 live in the per-lane register index; qubits 5-9 live in lane bits.
// Readout <Z_q> is a probs[16x1024] x sign[1024x16] GEMM done with
// v_wmma_f32_16x16x32_f16 per 16-wave workgroup.

typedef __attribute__((ext_vector_type(16))) _Float16 v16h;
typedef __attribute__((ext_vector_type(8)))  float    v8f;

#define NQ 10
#define NL 4
#define BATCH_N 4096
#define WAVES_PER_BLK 16
#define BLOCK_T (WAVES_PER_BLK * 32)
#define PI_F 3.14159265358979323846f

struct C2 {                    // 2x2 complex matrix, rr/ii = [row][col]
  float rr[2][2];
  float ii[2][2];
};

__device__ __forceinline__ C2 cmul(const C2& A, const C2& B) {
  C2 D;
#pragma unroll
  for (int i = 0; i < 2; ++i)
#pragma unroll
    for (int j = 0; j < 2; ++j) {
      float rr = 0.f, ii = 0.f;
#pragma unroll
      for (int k = 0; k < 2; ++k) {
        rr += A.rr[i][k] * B.rr[k][j] - A.ii[i][k] * B.ii[k][j];
        ii += A.rr[i][k] * B.ii[k][j] + A.ii[i][k] * B.rr[k][j];
      }
      D.rr[i][j] = rr; D.ii[i][j] = ii;
    }
  return D;
}

__global__ __launch_bounds__(BLOCK_T) void qcircuit_kernel(
    const float* __restrict__ x, const float* __restrict__ w,
    float* __restrict__ out)
{
  __shared__ _Float16 probs[WAVES_PER_BLK][1024];   // 32 KB

  const int lane = threadIdx.x & 31;
  const int wave = threadIdx.x >> 5;
  const int b    = blockIdx.x * WAVES_PER_BLK + wave;

  __builtin_prefetch(w, 0, 0);   // global_prefetch_b8: weights -> caches

  // Clipped half-angles of the data-encoding RY gates (reused every layer).
  float xh[NQ];
#pragma unroll
  for (int q = 0; q < NQ; ++q) {
    float v = x[b * NQ + q];
    v = fminf(fmaxf(v, -PI_F), PI_F);
    xh[q] = 0.5f * v;
  }

  // Statevector: amplitude index idx = (lane << 5) | r, qubit q = bit q of idx.
  float sr[32], si[32];
#pragma unroll
  for (int r = 0; r < 32; ++r) { sr[r] = 0.f; si[r] = 0.f; }
  sr[0] = (lane == 0) ? 1.f : 0.f;     // |0...0>

#pragma unroll 1
  for (int l = 0; l < NL; ++l) {
    const float* wl = w + l * NQ * 3;

    // ---- fused single-qubit gate V = RZ(w2)*RY(w1)*RX(w0)*RY(x) per qubit --
#pragma unroll
    for (int q = 0; q < NQ; ++q) {
      float se, ce; __sincosf(xh[q],              &se, &ce);
      float sx, cx; __sincosf(0.5f * wl[q*3 + 0], &sx, &cx);
      float sy, cy; __sincosf(0.5f * wl[q*3 + 1], &sy, &cy);
      float sz, cz; __sincosf(0.5f * wl[q*3 + 2], &sz, &cz);
      C2 Ue = { {{ce, -se}, {se,  ce}}, {{0.f, 0.f}, {0.f, 0.f}} };
      C2 Ux = { {{cx, 0.f}, {0.f, cx}}, {{0.f, -sx}, {-sx, 0.f}} };
      C2 Uy = { {{cy, -sy}, {sy,  cy}}, {{0.f, 0.f}, {0.f, 0.f}} };
      C2 Uz = { {{cz, 0.f}, {0.f, cz}}, {{-sz, 0.f}, {0.f,  sz}} };
      C2 V = cmul(Uz, cmul(Uy, cmul(Ux, Ue)));

      if (q < 5) {
        // Register-space qubit: pairs (r0, r0 | 1<<q) inside each lane.
#pragma unroll
        for (int r0 = 0; r0 < 32; ++r0) {
          if (r0 & (1 << q)) continue;
          const int r1 = r0 | (1 << q);
          float a0r = sr[r0], a0i = si[r0], a1r = sr[r1], a1i = si[r1];
          sr[r0] = V.rr[0][0]*a0r - V.ii[0][0]*a0i + V.rr[0][1]*a1r - V.ii[0][1]*a1i;
          si[r0] = V.rr[0][0]*a0i + V.ii[0][0]*a0r + V.rr[0][1]*a1i + V.ii[0][1]*a1r;
          sr[r1] = V.rr[1][0]*a0r - V.ii[1][0]*a0i + V.rr[1][1]*a1r - V.ii[1][1]*a1i;
          si[r1] = V.rr[1][0]*a0i + V.ii[1][0]*a0r + V.rr[1][1]*a1i + V.ii[1][1]*a1r;
        }
      } else {
        // Lane-space qubit: exchange partner amplitude across lanes.
        const int bp  = q - 5;
        const int bit = (lane >> bp) & 1;
        const float mr = bit ? V.rr[1][1] : V.rr[0][0];
        const float mi = bit ? V.ii[1][1] : V.ii[0][0];
        const float pr = bit ? V.rr[1][0] : V.rr[0][1];
        const float pim = bit ? V.ii[1][0] : V.ii[0][1];
#pragma unroll
        for (int r = 0; r < 32; ++r) {
          float qr = __shfl_xor(sr[r], 1 << bp, 32);
          float qi = __shfl_xor(si[r], 1 << bp, 32);
          float nr = mr * sr[r] - mi * si[r] + pr * qr - pim * qi;
          float ni = mr * si[r] + mi * sr[r] + pr * qi + pim * qr;
          sr[r] = nr; si[r] = ni;
        }
      }
    }

    // ---- CNOT ring (0,1)(1,2)...(8,9)(9,0) -------------------------------
    // (c, c+1) for c=0..3: both bits in register space -> free register swaps
#pragma unroll
    for (int c = 0; c < 4; ++c) {
      const int t = c + 1;
#pragma unroll
      for (int r = 0; r < 32; ++r) {
        if (((r >> c) & 1) == 1 && ((r >> t) & 1) == 0) {
          const int r2 = r | (1 << t);
          float tr = sr[r]; sr[r] = sr[r2]; sr[r2] = tr;
          float ti = si[r]; si[r] = si[r2]; si[r2] = ti;
        }
      }
    }
    // (4,5): control = reg bit 4, target = lane bit 0
#pragma unroll
    for (int r = 16; r < 32; ++r) {
      sr[r] = __shfl_xor(sr[r], 1, 32);
      si[r] = __shfl_xor(si[r], 1, 32);
    }
    // (5,6)..(8,9): both bits in lane space
#pragma unroll
    for (int c = 0; c < 4; ++c) {
      const int cb = (lane >> c) & 1;
      const int m  = 1 << (c + 1);
#pragma unroll
      for (int r = 0; r < 32; ++r) {
        float qr = __shfl_xor(sr[r], m, 32);
        float qi = __shfl_xor(si[r], m, 32);
        sr[r] = cb ? qr : sr[r];
        si[r] = cb ? qi : si[r];
      }
    }
    // (9,0): control = lane bit 4, target = reg bit 0
    {
      const int cb = (lane >> 4) & 1;
#pragma unroll
      for (int r = 0; r < 32; r += 2) {
        float t0r = sr[r], t0i = si[r], t1r = sr[r+1], t1i = si[r+1];
        sr[r]   = cb ? t1r : t0r;  si[r]   = cb ? t1i : t0i;
        sr[r+1] = cb ? t0r : t1r;  si[r+1] = cb ? t0i : t1i;
      }
    }
  }

  // ---- probabilities -> f16 -> LDS (64B-aligned b128 stores per lane) -----
  {
    unsigned int pk[16];
#pragma unroll
    for (int j = 0; j < 16; ++j) {
      float p0 = sr[2*j]   * sr[2*j]   + si[2*j]   * si[2*j];
      float p1 = sr[2*j+1] * sr[2*j+1] + si[2*j+1] * si[2*j+1];
      union { _Float16 h[2]; unsigned int u; } cvt;
      cvt.h[0] = (_Float16)p0; cvt.h[1] = (_Float16)p1;
      pk[j] = cvt.u;
    }
    uint4* d4 = (uint4*)&probs[wave][lane * 32];
#pragma unroll
    for (int j = 0; j < 4; ++j)
      d4[j] = make_uint4(pk[4*j], pk[4*j+1], pk[4*j+2], pk[4*j+3]);
  }
  __syncthreads();

  // ---- readout GEMM: D[16 batch x 16] = probs[16 x 1024] * sign[1024 x 16]
  // sign[k][n] = n<10 ? (1 - 2*bit_n(k)) : 0 ; computed, not loaded.
  if (wave == 0) {
    const int n  = lane & 15;
    const int hi = lane >> 4;
    v8f acc = {};
#pragma unroll 1
    for (int kc = 0; kc < 32; ++kc) {
      const int kb = kc * 32;
      // A fragment (16-bit A 16x32 layout): lanes<16 row=n K=0..7,16..23;
      // lanes>=16 row=n K=8..15,24..31.
      union { unsigned int u[8]; v16h h; } A;
      const unsigned int* prow = (const unsigned int*)&probs[n][0];
      const int koffA = kb + (hi ? 8 : 0);
#pragma unroll
      for (int j = 0; j < 4; ++j) {
        A.u[j]     = prow[(koffA + 2*j) >> 1];
        A.u[4 + j] = prow[(koffA + 16 + 2*j) >> 1];
      }
      // B fragment: lanes<16 hold K=kb+0..7,16..23 for col n; lanes>=16 K=kb+8..15,24..31.
      v16h B;
      const int kB = kb + (hi ? 8 : 0);
#pragma unroll
      for (int e = 0; e < 16; ++e) {
        const int k = kB + (e < 8 ? e : e + 8);
        float s = (n < NQ) ? (((k >> n) & 1) ? -1.0f : 1.0f) : 0.0f;
        B[e] = (_Float16)s;
      }
      acc = __builtin_amdgcn_wmma_f32_16x16x32_f16(
          /*neg_a=*/false, A.h, /*neg_b=*/false, B,
          /*c_mod=*/(short)0, acc, /*reuse_a=*/false, /*reuse_b=*/false);
    }
    // D layout: VGPR j -> M=j (lanes 0-15) / M=8+j (lanes 16-31), N=lane&15
    if (n < NQ) {
      const int base = blockIdx.x * WAVES_PER_BLK;
#pragma unroll
      for (int j = 0; j < 8; ++j) {
        const int m = (hi ? 8 : 0) + j;
        out[(base + m) * NQ + n] = acc[j];
      }
    }
  }
}

extern "C" void kernel_launch(void* const* d_in, const int* in_sizes, int n_in,
                              void* d_out, int out_size, void* d_ws, size_t ws_size,
                              hipStream_t stream) {
  const float* x = (const float*)d_in[0];   // [4096, 10] f32
  const float* w = (const float*)d_in[1];   // [4, 10, 3]  f32
  float* out = (float*)d_out;               // [4096, 10] f32
  dim3 grid(BATCH_N / WAVES_PER_BLK);
  dim3 block(BLOCK_T);
  hipLaunchKernelGGL(qcircuit_kernel, grid, block, 0, stream, x, w, out);
}
